// HybridQLSTM_65481071396300
// MI455X (gfx1250) — compile-verified, matrix operand order checked by reading, not run on hardware
//
#include <hip/hip_runtime.h>
#include <hip/hip_bf16.h>
#include <math.h>

// ---------------------------------------------------------------------------
// HybridQLSTM on MI455X (gfx1250).
//   - Per-step GEMM: Z(512x528) = [x_t, h_{t-1}](512x256) @ W_big^T, computed
//     with V_WMMA_F32_16X16X32_F16; one wave owns a 64x16 strip (4 WMMA accs,
//     B fragment reused 4x per K-step).
//   - Inputs pre-converted to f16 once (ws permitting) so the serial loop does
//     pure f16 b128 loads, no f32->f16 cvt on the critical path.
//   - qstep: 4-qubit statevector sim per gate + LSTM pointwise, fused.
// ---------------------------------------------------------------------------

typedef __attribute__((ext_vector_type(16))) _Float16 v16h;
typedef __attribute__((ext_vector_type(8)))  _Float16 v8h;
typedef __attribute__((ext_vector_type(4)))  _Float16 v4h;
typedef __attribute__((ext_vector_type(8)))  float    v8f;

#define TSTEPS 256
#define BATCH  512
#define DIN    128
#define HID    128
#define NQ     4
#define KDIM   (DIN + HID)        // 256
#define NOUT   (4 * HID + 4 * NQ) // 528
#define NTILES (NOUT / 16)        // 33
#define KSTEPS (KDIM / 32)        // 8

// Workspace layout (bytes; every section is a multiple of 256)
static constexpr size_t WPACK_HALFS = (size_t)NTILES * KSTEPS * 32 * 16;   // 135168
static constexpr size_t OFF_HX16    = WPACK_HALFS * 2;                     // 270336
static constexpr size_t OFF_CX      = OFF_HX16 + (size_t)BATCH * HID * 2;  // 401408
static constexpr size_t OFF_Z       = OFF_CX + (size_t)BATCH * HID * 4;    // 663552
static constexpr size_t OFF_XSTEP   = OFF_Z + (size_t)BATCH * NOUT * 4;    // 1744896
static constexpr size_t OFF_XALL    = OFF_XSTEP + (size_t)BATCH * DIN * 2; // 1875968
static constexpr size_t NEED_BIG    = OFF_XALL + (size_t)TSTEPS * BATCH * DIN * 2;

// ---------------------------------------------------------------------------
// Pack 528x256 fp32 weights -> f16, pre-swizzled into WMMA B-fragment lane
// order (mirror of the ISA 16-bit A layout with M<->N):
//   lane<16  : n = n0+lane,    K-octets {kk+0..7, kk+16..23}
//   lane>=16 : n = n0+lane-16, K-octets {kk+8..15, kk+24..31}
// Flat order [ntile][kstep][lane][elem] -> one contiguous v16h per lane/step.
// ---------------------------------------------------------------------------
__global__ __launch_bounds__(KDIM)
void qlstm_pack_weights(const float* __restrict__ Wf, const float* __restrict__ Wi,
                        const float* __restrict__ Wg, const float* __restrict__ Wo,
                        const float* __restrict__ Wfq, const float* __restrict__ Wiq,
                        const float* __restrict__ Wgq, const float* __restrict__ Woq,
                        _Float16* __restrict__ Wpack)
{
    const int n = blockIdx.x;   // 0..527
    const int k = threadIdx.x;  // 0..255

    float w;
    if (n < 4 * HID) {
        const int g = n >> 7, r = n & 127;
        const float* Wsel[4] = {Wf, Wi, Wg, Wo};
        w = Wsel[g][(size_t)r * KDIM + k];
    } else {
        const int m = n - 4 * HID;
        const int g = m >> 2, r = m & 3;
        const float* Wsel[4] = {Wfq, Wiq, Wgq, Woq};
        w = Wsel[g][(size_t)r * KDIM + k];
    }

    const int ntile = n >> 4, nl = n & 15;
    const int ks = k >> 5, ko = k & 31;
    const int half = ((ko >= 8 && ko < 16) || (ko >= 24)) ? 1 : 0;
    const int lane = nl + half * 16;
    int elem;
    if (!half) elem = (ko < 8)  ? ko       : (ko - 16) + 8;
    else       elem = (ko < 16) ? (ko - 8) : (ko - 24) + 8;

    Wpack[(((size_t)ntile * KSTEPS + ks) * 32 + lane) * 16 + elem] = (_Float16)w;
}

// ---------------------------------------------------------------------------
// f32 -> f16 vector convert (4 elems/thread, b128 in, b64 out).
// ---------------------------------------------------------------------------
__global__ __launch_bounds__(256)
void qlstm_cvt_f16(const float* __restrict__ in, _Float16* __restrict__ out)
{
    const size_t i = ((size_t)blockIdx.x * 256 + threadIdx.x) * 4;
    const float4 v = *reinterpret_cast<const float4*>(in + i);
    v4h o;
    o[0] = (_Float16)v.x; o[1] = (_Float16)v.y;
    o[2] = (_Float16)v.z; o[3] = (_Float16)v.w;
    *reinterpret_cast<v4h*>(out + i) = o;
}

// ---------------------------------------------------------------------------
// Initialize recurrent state from hx0 / cx0.
// ---------------------------------------------------------------------------
__global__ __launch_bounds__(256)
void qlstm_init(const float* __restrict__ hx0, const float* __restrict__ cx0,
                _Float16* __restrict__ hx16, float* __restrict__ cx)
{
    const int i = blockIdx.x * 256 + threadIdx.x;
    hx16[i] = (_Float16)hx0[i];
    cx[i]   = cx0[i];
}

// ---------------------------------------------------------------------------
// Z = comb @ W_big^T.  One wave computes a 64(M)x16(N) strip: 4 v8f
// accumulators, 8 K-steps, B fragment loaded once per K-step and reused by
// 4 WMMAs.  K-steps 0..3 read x16 (f16 inputs), 4..7 read hx16; the D=128
// split lands exactly on a K-step boundary.
// ---------------------------------------------------------------------------
__global__ __launch_bounds__(128)
void qlstm_gemm_step(const _Float16* __restrict__ x16,   // [BATCH, DIN] f16
                     const _Float16* __restrict__ hx16,  // [BATCH, HID] f16
                     const _Float16* __restrict__ Wpack, // packed B fragments
                     float* __restrict__ Z)              // [BATCH, NOUT]
{
    const int lane  = threadIdx.x & 31;
    const int wave  = threadIdx.x >> 5;
    const int ntile = blockIdx.x;                  // 0..32
    const int n0    = ntile * 16;
    const int m0    = (blockIdx.y * 4 + wave) * 64; // 64-row strip
    const int half  = lane >> 4;                    // K-octet group
    const int mlane = lane & 15;

    const _Float16* bp = Wpack + ((size_t)ntile * KSTEPS) * 32 * 16
                               + (size_t)lane * 16;

    v8f acc[4] = {v8f{}, v8f{}, v8f{}, v8f{}};

    // K 0..127 from x16
    #pragma unroll
    for (int ks = 0; ks < 4; ++ks) {
        const v16h b = *reinterpret_cast<const v16h*>(bp + (size_t)ks * 32 * 16);
        const int kb0 = ks * 32 + half * 8;
        const int kb1 = kb0 + 16;
        #pragma unroll
        for (int sub = 0; sub < 4; ++sub) {
            const _Float16* row = x16 + (size_t)(m0 + sub * 16 + mlane) * DIN;
            const v8h h0 = *reinterpret_cast<const v8h*>(row + kb0);
            const v8h h1 = *reinterpret_cast<const v8h*>(row + kb1);
            v16h a;
            #pragma unroll
            for (int e = 0; e < 8; ++e) { a[e] = h0[e]; a[8 + e] = h1[e]; }
            acc[sub] = __builtin_amdgcn_wmma_f32_16x16x32_f16(
                false, a, false, b, (short)0, acc[sub], false, false);
        }
    }
    // K 128..255 from hx16
    #pragma unroll
    for (int ks = 4; ks < 8; ++ks) {
        const v16h b = *reinterpret_cast<const v16h*>(bp + (size_t)ks * 32 * 16);
        const int kb0 = (ks - 4) * 32 + half * 8;
        const int kb1 = kb0 + 16;
        #pragma unroll
        for (int sub = 0; sub < 4; ++sub) {
            const _Float16* row = hx16 + (size_t)(m0 + sub * 16 + mlane) * HID;
            const v8h h0 = *reinterpret_cast<const v8h*>(row + kb0);
            const v8h h1 = *reinterpret_cast<const v8h*>(row + kb1);
            v16h a;
            #pragma unroll
            for (int e = 0; e < 8; ++e) { a[e] = h0[e]; a[8 + e] = h1[e]; }
            acc[sub] = __builtin_amdgcn_wmma_f32_16x16x32_f16(
                false, a, false, b, (short)0, acc[sub], false, false);
        }
    }

    // C/D layout: VGPR i, lanes 0-15 -> row +i, lanes 16-31 -> row +8+i,
    // column n0 + (lane & 15).
    const int col = n0 + mlane;
    #pragma unroll
    for (int sub = 0; sub < 4; ++sub) {
        #pragma unroll
        for (int i = 0; i < 8; ++i)
            Z[(size_t)(m0 + sub * 16 + i + half * 8) * NOUT + col] = acc[sub][i];
    }
}

// ---------------------------------------------------------------------------
// 4-qubit statevector helpers (16 complex amps in registers, fully unrolled).
// Wire w has bit-stride 8>>w (big-endian, matching the reference reshape).
// ---------------------------------------------------------------------------
__device__ __forceinline__
void rx_layer(float* sr, float* si, const float* ang)
{
    #pragma unroll
    for (int w = 0; w < 4; ++w) {
        const float hh = 0.5f * ang[w];
        const float c = cosf(hh), s = sinf(hh);   // RX: [[c, -i s],[-i s, c]]
        const int st = 8 >> w;
        #pragma unroll
        for (int base = 0; base < 16; ++base) {
            if (base & st) continue;
            const int p1 = base + st;
            const float r0 = sr[base], i0 = si[base];
            const float r1 = sr[p1],   i1 = si[p1];
            sr[base] = c * r0 + s * i1;
            si[base] = c * i0 - s * r1;
            sr[p1]   = s * i0 + c * r1;
            si[p1]   = c * i1 - s * r0;
        }
    }
}

// ---------------------------------------------------------------------------
// Per-step fused: quantum gates (threads 0..3), readout, LSTM pointwise.
// One block per batch row, HID threads.
// ---------------------------------------------------------------------------
__global__ __launch_bounds__(HID)
void qlstm_qstep(const float* __restrict__ Z,
                 float* __restrict__ cx,
                 _Float16* __restrict__ hx16,
                 float* __restrict__ out_t,
                 const float* __restrict__ bf, const float* __restrict__ bi,
                 const float* __restrict__ bg, const float* __restrict__ bo,
                 const float* __restrict__ bfq, const float* __restrict__ biq,
                 const float* __restrict__ bgq, const float* __restrict__ boq,
                 const float* __restrict__ thf, const float* __restrict__ thi,
                 const float* __restrict__ thg, const float* __restrict__ tho,
                 const float* __restrict__ Wq, const float* __restrict__ bq,
                 float* __restrict__ hxT_out, float* __restrict__ cxT_out)
{
    __shared__ float expz[4][4];   // [gate][qubit]
    const int b = blockIdx.x;
    const int h = threadIdx.x;
    const float* Zrow = Z + (size_t)b * NOUT;

    if (h < 4) {
        const int g = h;
        const float* bqv[4] = {bfq, biq, bgq, boq};
        const float* thv[4] = {thf, thi, thg, tho};
        float ang[4], th[4];
        #pragma unroll
        for (int i = 0; i < 4; ++i) {
            ang[i] = Zrow[4 * HID + g * 4 + i] + bqv[g][i];
            th[i]  = thv[g][i];
        }
        float sr[16], si[16];
        #pragma unroll
        for (int p = 0; p < 16; ++p) { sr[p] = 0.f; si[p] = 0.f; }
        sr[0] = 1.f;

        rx_layer(sr, si, ang);                 // RX(angles) on wires 0..3
        #pragma unroll
        for (int w = 0; w < 3; ++w) {          // CNOT(ctrl=w, tgt=w+1)
            const int sc = 8 >> w, st = 4 >> w;
            #pragma unroll
            for (int base = 0; base < 16; ++base) {
                if ((base & sc) && !(base & st)) {
                    const float tr = sr[base], ti = si[base];
                    sr[base] = sr[base + st]; si[base] = si[base + st];
                    sr[base + st] = tr;       si[base + st] = ti;
                }
            }
        }
        rx_layer(sr, si, th);                  // RX(theta) on wires 0..3

        #pragma unroll
        for (int i = 0; i < 4; ++i) {          // <Z_i> = P(bit=0) - P(bit=1)
            const int msk = 8 >> i;
            float e = 0.f;
            #pragma unroll
            for (int p = 0; p < 16; ++p) {
                const float pr = sr[p] * sr[p] + si[p] * si[p];
                e += (p & msk) ? -pr : pr;
            }
            expz[g][i] = e;
        }
    }
    __syncthreads();

    const float wq0 = Wq[h * 4 + 0], wq1 = Wq[h * 4 + 1];
    const float wq2 = Wq[h * 4 + 2], wq3 = Wq[h * 4 + 3];
    const float bqh = bq[h];
    const float* bvec[4] = {bf, bi, bg, bo};

    float gate[4];
    #pragma unroll
    for (int n = 0; n < 4; ++n) {
        const float lin = Zrow[n * HID + h] + bvec[n][h];
        const float qv  = expz[n][0] * wq0 + expz[n][1] * wq1 +
                          expz[n][2] * wq2 + expz[n][3] * wq3 + bqh;
        if (n == 2) gate[n] = tanhf(lin) + tanhf(qv);
        else        gate[n] = 1.f / (1.f + expf(-lin)) + 1.f / (1.f + expf(-qv));
    }

    const size_t idx = (size_t)b * HID + h;
    const float cx2 = gate[0] * cx[idx] + gate[1] * gate[2];
    const float hx2 = gate[3] * tanhf(cx2);
    cx[idx]   = cx2;
    hx16[idx] = (_Float16)hx2;
    out_t[idx] = hx2;
    if (hxT_out) { hxT_out[idx] = hx2; cxT_out[idx] = cx2; }
}

// ---------------------------------------------------------------------------
// Host launcher.  Pre-converts all inputs to f16 when ws_size allows (the
// choice depends only on ws_size -> deterministic); otherwise converts one
// step's x slice per iteration.  256 sequential (gemm, qstep) pairs.
// ---------------------------------------------------------------------------
extern "C" void kernel_launch(void* const* d_in, const int* in_sizes, int n_in,
                              void* d_out, int out_size, void* d_ws, size_t ws_size,
                              hipStream_t stream)
{
    const float* inputs = (const float*)d_in[0];
    const float* hx0    = (const float*)d_in[1];
    const float* cx0    = (const float*)d_in[2];
    const float* Wf  = (const float*)d_in[3];  const float* bf  = (const float*)d_in[4];
    const float* Wfq = (const float*)d_in[5];  const float* bfq = (const float*)d_in[6];
    const float* thf = (const float*)d_in[7];
    const float* Wi  = (const float*)d_in[8];  const float* bi  = (const float*)d_in[9];
    const float* Wiq = (const float*)d_in[10]; const float* biq = (const float*)d_in[11];
    const float* thi = (const float*)d_in[12];
    const float* Wg  = (const float*)d_in[13]; const float* bg  = (const float*)d_in[14];
    const float* Wgq = (const float*)d_in[15]; const float* bgq = (const float*)d_in[16];
    const float* thg = (const float*)d_in[17];
    const float* Wo  = (const float*)d_in[18]; const float* bo  = (const float*)d_in[19];
    const float* Woq = (const float*)d_in[20]; const float* boq = (const float*)d_in[21];
    const float* tho = (const float*)d_in[22];
    const float* Wq  = (const float*)d_in[23]; const float* bq  = (const float*)d_in[24];

    char* ws = (char*)d_ws;
    _Float16* Wpack  = (_Float16*)ws;
    _Float16* hx16   = (_Float16*)(ws + OFF_HX16);
    float*    cx     = (float*)(ws + OFF_CX);
    float*    Z      = (float*)(ws + OFF_Z);
    _Float16* xstep  = (_Float16*)(ws + OFF_XSTEP);
    _Float16* xall   = (_Float16*)(ws + OFF_XALL);
    float*    outs   = (float*)d_out;

    const bool big = (ws_size >= NEED_BIG);

    qlstm_pack_weights<<<dim3(NOUT), dim3(KDIM), 0, stream>>>(
        Wf, Wi, Wg, Wo, Wfq, Wiq, Wgq, Woq, Wpack);
    qlstm_init<<<dim3((BATCH * HID) / 256), dim3(256), 0, stream>>>(hx0, cx0, hx16, cx);
    if (big) {
        // Convert the whole (T,B,D) input tensor to f16 once, in parallel.
        qlstm_cvt_f16<<<dim3((TSTEPS * BATCH * DIN) / 1024), dim3(256), 0, stream>>>(
            inputs, xall);
    }

    float* hxT = outs + (size_t)TSTEPS * BATCH * HID;
    float* cxT = hxT + (size_t)BATCH * HID;

    for (int t = 0; t < TSTEPS; ++t) {
        const _Float16* x16_t;
        if (big) {
            x16_t = xall + (size_t)t * BATCH * DIN;
        } else {
            qlstm_cvt_f16<<<dim3((BATCH * DIN) / 1024), dim3(256), 0, stream>>>(
                inputs + (size_t)t * BATCH * DIN, xstep);
            x16_t = xstep;
        }
        qlstm_gemm_step<<<dim3(NTILES, 2), dim3(128), 0, stream>>>(
            x16_t, hx16, Wpack, Z);
        qlstm_qstep<<<dim3(BATCH), dim3(HID), 0, stream>>>(
            Z, cx, hx16, outs + (size_t)t * BATCH * HID,
            bf, bi, bg, bo, bfq, biq, bgq, boq,
            thf, thi, thg, tho, Wq, bq,
            (t == TSTEPS - 1) ? hxT : nullptr,
            (t == TSTEPS - 1) ? cxT : nullptr);
    }
    (void)in_sizes; (void)n_in; (void)out_size;
}
